// CTM_ASR_Model_87797721465456
// MI455X (gfx1250) — compile-verified, matrix operand order checked by reading, not run on hardware
//
#include <hip/hip_runtime.h>
#include <hip/hip_bf16.h>

// ---------------------------------------------------------------------------
// CTM ASR scan for MI455X (gfx1250) — persistent single-workgroup design.
//
// 8 wave32 waves (256 threads): 2 waves/SIMD so each wave may hold ~512 VGPRs
// without spilling. Each wave keeps its 64 columns of Wf (4 N-tiles x 10
// K-slices) + Wu (4 tiles) as persistent bf16 WMMA B-fragments in VGPRs.
// Wd/Wb/Wh B-fragments and all activations live in LDS (~288 KB of the 320 KB
// WGP budget). The st trace is a 10-deep circular bf16 ring; ast collapses to
// its last column. Logits are computed with 17 chained WMMAs on wave 0.
// ---------------------------------------------------------------------------

typedef __attribute__((ext_vector_type(16))) __bf16 v16bf;
typedef __attribute__((ext_vector_type(8)))  float  v8f;

namespace {
constexpr int kB = 16, kT = 1500, kDin = 64, kD = 256, kMem = 10;
constexpr int kSyncDim = 528, kVocab = 15, kMinw = 16;
constexpr int NT = 256;           // 8 waves

// dynamic-LDS layout (bytes, 32B aligned)
constexpr int OFF_ST      = 0;       // bf16 [16][256][10] trace ring      (81920)
constexpr int OFF_H0FULL  = 81920;   // f32  [16][512] GEMM1/3 staging     (32768)
constexpr int OFF_PREFRAG = 114688;  // bf16 10x512 pre A-fragments        (10240)
constexpr int OFF_XFRAG   = 124928;  // bf16 2x512  x_t A-fragments        (2048)
constexpr int OFF_H0FRAG  = 126976;  // bf16 8x512  h0 A-fragments         (8192)
constexpr int OFF_D1FRAG  = 135168;  // bf16 1x512  d1 A-frag (K padded)   (1024)
constexpr int OFF_H0F32   = 136192;  // f32  [16][256] h0 residual         (16384)
constexpr int OFF_ACT     = 152576;  // f32  [16][256] current act         (16384)
constexpr int OFF_D1PRE   = 168960;  // f32  [16][32]                      (2048)
constexpr int OFF_SYNCF   = 171008;  // bf16 17x512 sync A-frags (K pad)   (17408)
constexpr int OFF_WHF     = 188416;  // bf16 17x512 Wh B-frags (pads)      (17408)
constexpr int OFF_LUT     = 205824;  // u32  [528] (i<<8|j) triu pairs     (2112)
constexpr int OFF_BFL     = 207936;  // f32 [512]
constexpr int OFF_BUL     = 209984;  // f32 [512]
constexpr int OFF_BBL     = 212032;  // f32 [64]
constexpr int OFF_BDL     = 212288;  // f32 [32]
constexpr int OFF_GFL     = 212416;  // f32 [256] x6 LN params
constexpr int OFF_BEFL    = 213440;
constexpr int OFF_GUL     = 214464;
constexpr int OFF_BEUL    = 215488;
constexpr int OFF_GSL     = 216512;
constexpr int OFF_BESL    = 217536;
constexpr int OFF_GDL     = 218560;  // f32 [16]
constexpr int OFF_BEDL    = 218624;  // f32 [16]
constexpr int OFF_BHL     = 218688;  // f32 [15]
constexpr int OFF_WDF     = 218752;  // bf16 2 tiles x 8 slices x 512      (16384)
constexpr int OFF_WBF     = 235136;  // bf16 4 tiles x 2 slices x 512      (8192)
constexpr int OFF_W1L     = 243328;  // f32 [10][256][4]                   (40960)
constexpr int OFF_B1L     = 284288;  // f32 [256][4]                       (4096)
constexpr int OFF_W2L     = 288384;  // f32 [2][256][2]                    (4096)
constexpr int OFF_B2L     = 292480;  // f32 [256][2]                       (2048)
constexpr int SMEM_BYTES  = 294528;  // ~287.6 KB < 320 KB/WGP
} // namespace

__device__ __forceinline__ float sigm(float x) { return 1.0f / (1.0f + __expf(-x)); }

__device__ __forceinline__ float wred(float v) {
  v += __shfl_xor(v, 16, 32);
  v += __shfl_xor(v, 8, 32);
  v += __shfl_xor(v, 4, 32);
  v += __shfl_xor(v, 2, 32);
  v += __shfl_xor(v, 1, 32);
  return v;
}

// Fragment layout: element (m,k) of a 16xK A-matrix (or (k,n) of B) lives at
// slice s=k/32, lane=(k%32)/16*16 + m, elem e=k%16. Each lane's 16 bf16
// operand values are contiguous (32B) -> one wide ds_load per fragment.
__device__ __forceinline__ void fstore(__bf16* frag, int s, int m, int kp, float v) {
  frag[s * 512 + ((kp >> 4) * 16 + m) * 16 + (kp & 15)] = (__bf16)v;
}

__device__ __forceinline__ v8f wmma_bf16(v16bf a, v16bf b, v8f c) {
  return __builtin_amdgcn_wmma_f32_16x16x32_bf16(false, a, false, b, (short)0, c,
                                                 false, false);
}

__global__ void __launch_bounds__(NT, 1)
ctm_asr_kernel(const float* __restrict__ X,   const float* __restrict__ st0,
               const float* __restrict__ ast0,
               const float* __restrict__ Wb,  const float* __restrict__ bb,
               const float* __restrict__ Wf,  const float* __restrict__ bf,
               const float* __restrict__ gf,  const float* __restrict__ bef,
               const float* __restrict__ Wd,  const float* __restrict__ bd,
               const float* __restrict__ gd,  const float* __restrict__ bed,
               const float* __restrict__ Wu,  const float* __restrict__ bu,
               const float* __restrict__ gu,  const float* __restrict__ beu,
               const float* __restrict__ gs,  const float* __restrict__ bes,
               const float* __restrict__ w1,  const float* __restrict__ b1,
               const float* __restrict__ w2,  const float* __restrict__ b2,
               const float* __restrict__ Wh,  const float* __restrict__ bh,
               float* __restrict__ out)
{
  extern __shared__ char smem[];
  __bf16* stBuf    = (__bf16*)(smem + OFF_ST);
  float*  h0full   = (float*)(smem + OFF_H0FULL);
  __bf16* preFrag  = (__bf16*)(smem + OFF_PREFRAG);
  __bf16* xFrag    = (__bf16*)(smem + OFF_XFRAG);
  __bf16* h0Frag   = (__bf16*)(smem + OFF_H0FRAG);
  __bf16* d1Frag   = (__bf16*)(smem + OFF_D1FRAG);
  float*  h0f32    = (float*)(smem + OFF_H0F32);
  float*  actBuf   = (float*)(smem + OFF_ACT);
  float*  d1pre    = (float*)(smem + OFF_D1PRE);
  __bf16* syncFrag = (__bf16*)(smem + OFF_SYNCF);
  __bf16* WhF      = (__bf16*)(smem + OFF_WHF);
  unsigned* pairLUT = (unsigned*)(smem + OFF_LUT);
  float* bfL  = (float*)(smem + OFF_BFL);
  float* buL  = (float*)(smem + OFF_BUL);
  float* bbL  = (float*)(smem + OFF_BBL);
  float* bdL  = (float*)(smem + OFF_BDL);
  float* gfL  = (float*)(smem + OFF_GFL);
  float* befL = (float*)(smem + OFF_BEFL);
  float* guL  = (float*)(smem + OFF_GUL);
  float* beuL = (float*)(smem + OFF_BEUL);
  float* gsL  = (float*)(smem + OFF_GSL);
  float* besL = (float*)(smem + OFF_BESL);
  float* gdL  = (float*)(smem + OFF_GDL);
  float* bedL = (float*)(smem + OFF_BEDL);
  float* bhL  = (float*)(smem + OFF_BHL);
  __bf16* WDF = (__bf16*)(smem + OFF_WDF);
  __bf16* WBF = (__bf16*)(smem + OFF_WBF);
  float* w1L = (float*)(smem + OFF_W1L);
  float* b1L = (float*)(smem + OFF_B1L);
  float* w2L = (float*)(smem + OFF_W2L);
  float* b2L = (float*)(smem + OFF_B2L);

  const int tid  = threadIdx.x;
  const int lane = tid & 31;
  const int w    = tid >> 5;     // wave id 0..7
  const int lcol = lane & 15;
  const int hi   = lane >> 4;
  const int colbase = w * 64;    // this wave's 64 output columns (4 N-tiles)

  // ------------------------- one-time setup ------------------------------
  for (int i = tid; i < 512; i += NT) { bfL[i] = bf[i]; buL[i] = bu[i]; }
  for (int i = tid; i < 256; i += NT) {
    gfL[i] = gf[i]; befL[i] = bef[i]; guL[i] = gu[i];
    beuL[i] = beu[i]; gsL[i] = gs[i]; besL[i] = bes[i];
  }
  if (tid < 64) bbL[tid] = bb[tid];
  if (tid < 32) bdL[tid] = bd[tid];
  if (tid < 16) { gdL[tid] = gd[tid]; bedL[tid] = bed[tid]; }
  if (tid < 15) bhL[tid] = bh[tid];
  if (tid < 32) {  // row-major upper-triangle pair LUT
    int i = tid, base = 32 * i - (i * (i - 1)) / 2;
    for (int j = i; j < 32; ++j) pairLUT[base + (j - i)] = (unsigned)((i << 8) | j);
  }
  // NLM weights into LDS
  for (int i = tid; i < kMem * kD * 4; i += NT) w1L[i] = w1[i];
  for (int i = tid; i < kD * 4; i += NT)        b1L[i] = b1[i];
  for (int i = tid; i < 2 * kD * 2; i += NT)    w2L[i] = w2[i];
  for (int i = tid; i < kD * 2; i += NT)        b2L[i] = b2[i];
  // trace init: broadcast start_trace; slot 9 == newest (m = 9)
  for (int i = tid; i < kB * kD * kMem; i += NT) {
    int b = i / (kD * kMem), r = i % (kD * kMem), d = r / kMem, m = r % kMem;
    stBuf[(b * kD + d) * kMem + m] = (__bf16)st0[d * kMem + m];
  }
  // act init = ast0[:, -1]; also stage into pre A-fragments (cols 64..319)
  for (int i = tid; i < kB * kD; i += NT) {
    int b = i >> 8, d = i & 255;
    float v = ast0[d * kMem + (kMem - 1)];
    actBuf[b * kD + d] = v;
    fstore(preFrag, 2 + (d >> 5), b, d & 31, v);
  }
  for (int i = tid; i < 512; i += NT) d1Frag[i] = (__bf16)0.0f;  // K-pad stays 0
  for (int i = tid; i < 17 * 512; i += NT) syncFrag[i] = (__bf16)0.0f; // pads stay 0
  // Wd (2 tiles x 8 slices) and Wb (4 tiles x 2 slices) B-fragments in LDS
  for (int i = tid; i < 8192; i += NT) {
    int tw = i / 4096, r = i % 4096, s = r / 512, l2 = r % 512;
    int ln = l2 / 16, e = l2 % 16;
    int k = s * 32 + (ln >> 4) * 16 + e, n = tw * 16 + (ln & 15);
    WDF[i] = (__bf16)Wd[k * 32 + n];
  }
  for (int i = tid; i < 4096; i += NT) {
    int tw = i / 1024, r = i % 1024, s = r / 512, l2 = r % 512;
    int ln = l2 / 16, e = l2 % 16;
    int k = s * 32 + (ln >> 4) * 16 + e, n = tw * 16 + (ln & 15);
    WBF[i] = (__bf16)Wb[k * 64 + n];
  }
  // Wh B-fragments: K padded 528->544, N padded 15->16
  for (int i = tid; i < 17 * 512; i += NT) {
    int s = i / 512, l2 = i % 512, ln = l2 / 16, e = l2 % 16;
    int k = s * 32 + (ln >> 4) * 16 + e, n = ln & 15;
    WhF[i] = (__bf16)((k < kSyncDim && n < kVocab) ? Wh[k * kVocab + n] : 0.0f);
  }

  // Persistent register B-fragments: Wf (4 tiles x 10 K-slices), Wu (4 tiles)
  v16bf wfr[4][10];
#pragma unroll
  for (int tt = 0; tt < 4; ++tt) {
#pragma unroll
    for (int s = 0; s < 10; ++s) {
      int n = colbase + tt * 16 + lcol;
#pragma unroll
      for (int e = 0; e < 16; ++e) {
        int k = s * 32 + hi * 16 + e;
        wfr[tt][s][e] = (__bf16)Wf[k * 512 + n];
      }
    }
  }
  v16bf wur[4];
#pragma unroll
  for (int tt = 0; tt < 4; ++tt) {
    int n = colbase + tt * 16 + lcol;
#pragma unroll
    for (int e = 0; e < 16; ++e) {
      int k = hi * 16 + e;
      wur[tt][e] = (__bf16)((k < kMinw) ? Wu[k * 512 + n] : 0.0f);
    }
  }

  __syncthreads();

  int slot = kMem - 1;  // index of newest trace entry

  // =========================== sequential scan ===========================
#pragma unroll 1
  for (int t = 0; t < kT; ++t) {
    // stage x_t into A-fragment layout
    for (int i = tid; i < kB * kDin; i += NT) {
      int b = i >> 6, ii = i & 63;
      fstore(xFrag, ii >> 5, b, ii & 31, X[(b * kT + t) * kDin + ii]);
    }
    __syncthreads();
    // kv = relu(x_t @ Wb + bb) -> pre columns [0,64); waves 0..3, 1 tile each
    if (w < 4) {
      int n0 = w * 16 + lcol;
      float bv = bbL[n0];
      v8f acc;
#pragma unroll
      for (int r = 0; r < 8; ++r) acc[r] = bv;
#pragma unroll
      for (int s = 0; s < 2; ++s) {
        v16bf a  = *(const v16bf*)(xFrag + s * 512 + lane * 16);
        v16bf bm = *(const v16bf*)(WBF + (w * 2 + s) * 512 + lane * 16);
        acc = wmma_bf16(a, bm, acc);
      }
#pragma unroll
      for (int r = 0; r < 8; ++r) {
        int m = r + 8 * hi;
        float v = acc[r]; v = v > 0.0f ? v : 0.0f;
        fstore(preFrag, n0 >> 5, m, n0 & 31, v);
      }
    }
    __syncthreads();

#pragma unroll 1
    for (int it = 0; it < 2; ++it) {
      // ---- GEMM1: h0pre = pre @ Wf + bf  (16x512, K=320; 40 WMMA/wave)
      {
        v8f ac[4];
#pragma unroll
        for (int tt = 0; tt < 4; ++tt) {
          float bv = bfL[colbase + tt * 16 + lcol];
#pragma unroll
          for (int r = 0; r < 8; ++r) ac[tt][r] = bv;
        }
#pragma unroll
        for (int s = 0; s < 10; ++s) {
          v16bf a = *(const v16bf*)(preFrag + s * 512 + lane * 16);
#pragma unroll
          for (int tt = 0; tt < 4; ++tt) ac[tt] = wmma_bf16(a, wfr[tt][s], ac[tt]);
        }
#pragma unroll
        for (int tt = 0; tt < 4; ++tt) {
          int n = colbase + tt * 16 + lcol;
#pragma unroll
          for (int r = 0; r < 8; ++r) h0full[(r + 8 * hi) * 512 + n] = ac[tt][r];
        }
      }
      __syncthreads();
      // ---- GLU + LN -> h0 ; wave w owns rows w and w+8
#pragma unroll
      for (int half = 0; half < 2; ++half) {
        int b = w + 8 * half;
        float v[8], s1 = 0.f, s2 = 0.f;
#pragma unroll
        for (int j = 0; j < 8; ++j) {
          int d = lane * 8 + j;
          float x = h0full[b * 512 + d] * sigm(h0full[b * 512 + 256 + d]);
          v[j] = x; s1 += x; s2 += x * x;
        }
        s1 = wred(s1); s2 = wred(s2);
        float mean = s1 * (1.f / 256.f);
        float inv  = rsqrtf(s2 * (1.f / 256.f) - mean * mean + 1e-5f);
#pragma unroll
        for (int j = 0; j < 8; ++j) {
          int d = lane * 8 + j;
          float hn = (v[j] - mean) * inv * gfL[d] + befL[d];
          h0f32[b * 256 + d] = hn;
          fstore(h0Frag, d >> 5, b, d & 31, hn);
        }
      }
      __syncthreads();
      // ---- GEMM2: d1pre = h0 @ Wd + bd  (16x32, K=256); waves 0..1
      if (w < 2) {
        int n0 = w * 16 + lcol;
        float bv = bdL[n0];
        v8f acc;
#pragma unroll
        for (int r = 0; r < 8; ++r) acc[r] = bv;
#pragma unroll
        for (int s = 0; s < 8; ++s) {
          v16bf a  = *(const v16bf*)(h0Frag + s * 512 + lane * 16);
          v16bf bm = *(const v16bf*)(WDF + (w * 8 + s) * 512 + lane * 16);
          acc = wmma_bf16(a, bm, acc);
        }
#pragma unroll
        for (int r = 0; r < 8; ++r) d1pre[(r + 8 * hi) * 32 + n0] = acc[r];
      }
      __syncthreads();
      // ---- GLU + LN -> d1 (16x16) ; wave 0, lane l owns row l
      if (w == 0 && lane < 16) {
        int b = lane;
        float v[16], s1 = 0.f, s2 = 0.f;
#pragma unroll
        for (int j = 0; j < 16; ++j) {
          float x = d1pre[b * 32 + j] * sigm(d1pre[b * 32 + 16 + j]);
          v[j] = x; s1 += x; s2 += x * x;
        }
        float mean = s1 * (1.f / 16.f);
        float inv  = rsqrtf(s2 * (1.f / 16.f) - mean * mean + 1e-5f);
#pragma unroll
        for (int j = 0; j < 16; ++j)
          fstore(d1Frag, 0, b, j, (v[j] - mean) * inv * gdL[j] + bedL[j]);
      }
      __syncthreads();
      // ---- GEMM3: u0pre = d1 @ Wu + bu  (16x512, K padded 16->32)
      {
        v8f ac[4];
#pragma unroll
        for (int tt = 0; tt < 4; ++tt) {
          float bv = buL[colbase + tt * 16 + lcol];
#pragma unroll
          for (int r = 0; r < 8; ++r) ac[tt][r] = bv;
        }
        v16bf a = *(const v16bf*)(d1Frag + lane * 16);
#pragma unroll
        for (int tt = 0; tt < 4; ++tt) ac[tt] = wmma_bf16(a, wur[tt], ac[tt]);
#pragma unroll
        for (int tt = 0; tt < 4; ++tt) {
          int n = colbase + tt * 16 + lcol;
#pragma unroll
          for (int r = 0; r < 8; ++r) h0full[(r + 8 * hi) * 512 + n] = ac[tt][r];
        }
      }
      __syncthreads();
      // ---- GLU+LN(u0) + residual + LN -> state ; push into trace ring
      int nsl = slot + 1; if (nsl >= kMem) nsl = 0;
#pragma unroll
      for (int half = 0; half < 2; ++half) {
        int b = w + 8 * half;
        float v[8], s1 = 0.f, s2 = 0.f;
#pragma unroll
        for (int j = 0; j < 8; ++j) {
          int d = lane * 8 + j;
          float x = h0full[b * 512 + d] * sigm(h0full[b * 512 + 256 + d]);
          v[j] = x; s1 += x; s2 += x * x;
        }
        s1 = wred(s1); s2 = wred(s2);
        float mean = s1 * (1.f / 256.f);
        float inv  = rsqrtf(s2 * (1.f / 256.f) - mean * mean + 1e-5f);
        float t1 = 0.f, t2 = 0.f;
#pragma unroll
        for (int j = 0; j < 8; ++j) {
          int d = lane * 8 + j;
          float sv = (v[j] - mean) * inv * guL[d] + beuL[d] + h0f32[b * 256 + d];
          v[j] = sv; t1 += sv; t2 += sv * sv;
        }
        t1 = wred(t1); t2 = wred(t2);
        float m2   = t1 * (1.f / 256.f);
        float inv2 = rsqrtf(t2 * (1.f / 256.f) - m2 * m2 + 1e-5f);
#pragma unroll
        for (int j = 0; j < 8; ++j) {
          int d = lane * 8 + j;
          stBuf[(b * 256 + d) * kMem + nsl] =
              (__bf16)((v[j] - m2) * inv2 * gsL[d] + besL[d]);
        }
      }
      __syncthreads();
      slot = nsl;
      // ---- NLM: per-(b,d) GLU chain over the 10-deep trace; thread owns d=tid
      {
        int idxs[10];
        int base = slot + 1; if (base >= kMem) base = 0;
#pragma unroll
        for (int i = 0; i < 10; ++i) { int s_ = base + i; if (s_ >= kMem) s_ -= kMem; idxs[i] = s_; }
        const int d = tid;
        float w1v[10][4];
#pragma unroll
        for (int m = 0; m < 10; ++m)
#pragma unroll
          for (int h = 0; h < 4; ++h) w1v[m][h] = w1L[(m * kD + d) * 4 + h];
        float b1v0 = b1L[d * 4 + 0], b1v1 = b1L[d * 4 + 1];
        float b1v2 = b1L[d * 4 + 2], b1v3 = b1L[d * 4 + 3];
        float w200 = w2L[d * 2 + 0], w201 = w2L[d * 2 + 1];
        float w210 = w2L[(kD + d) * 2 + 0], w211 = w2L[(kD + d) * 2 + 1];
        float b2v0 = b2L[d * 2 + 0], b2v1 = b2L[d * 2 + 1];
#pragma unroll 1
        for (int b = 0; b < kB; ++b) {
          const __bf16* sp = stBuf + (b * 256 + d) * kMem;
          float hv0 = b1v0, hv1 = b1v1, hv2 = b1v2, hv3 = b1v3;
#pragma unroll
          for (int i = 0; i < 10; ++i) {
            float sv = (float)sp[idxs[i]];
            hv0 += sv * w1v[i][0]; hv1 += sv * w1v[i][1];
            hv2 += sv * w1v[i][2]; hv3 += sv * w1v[i][3];
          }
          float g0 = hv0 * sigm(hv2), g1 = hv1 * sigm(hv3);
          float o0 = b2v0 + g0 * w200 + g1 * w210;
          float o1 = b2v1 + g0 * w201 + g1 * w211;
          float a = o0 * sigm(o1);
          actBuf[b * 256 + d] = a;
          fstore(preFrag, 2 + (d >> 5), b, d & 31, a);  // next iteration's pre
        }
      }
      __syncthreads();
    }  // iters

    // ---- sync features into A-fragments; logits via WMMA on wave 0 ----
    for (int i = tid; i < kB * kSyncDim; i += NT) {
      int b = i / kSyncDim, p = i - b * kSyncDim;
      unsigned u = pairLUT[p];
      float sv = actBuf[b * 256 + (int)(u >> 8)] * actBuf[b * 256 + (int)(u & 255u)];
      fstore(syncFrag, p >> 5, b, p & 31, sv);
    }
    __syncthreads();
    if (w == 0) {  // logits = sync @ Wh + bh : 17 chained K=32 WMMAs
      v8f acc;
      float bv = (lcol < kVocab) ? bhL[lcol] : 0.0f;
#pragma unroll
      for (int r = 0; r < 8; ++r) acc[r] = bv;
#pragma unroll
      for (int s = 0; s < 17; ++s) {
        v16bf a  = *(const v16bf*)(syncFrag + s * 512 + lane * 16);
        v16bf bm = *(const v16bf*)(WhF + s * 512 + lane * 16);
        acc = wmma_bf16(a, bm, acc);
      }
      if (lcol < kVocab) {
#pragma unroll
        for (int r = 0; r < 8; ++r) {
          int b = r + 8 * hi;
          out[(b * kT + t) * kVocab + lcol] = acc[r];
        }
      }
    }
    __syncthreads();
  }  // t
}

extern "C" void kernel_launch(void* const* d_in, const int* in_sizes, int n_in,
                              void* d_out, int out_size, void* d_ws, size_t ws_size,
                              hipStream_t stream) {
  (void)in_sizes; (void)n_in; (void)out_size; (void)d_ws; (void)ws_size;
  const float* X   = (const float*)d_in[0];
  const float* st0 = (const float*)d_in[1];
  const float* as0 = (const float*)d_in[2];
  const float* Wb  = (const float*)d_in[3];
  const float* bb  = (const float*)d_in[4];
  const float* Wf  = (const float*)d_in[5];
  const float* bf  = (const float*)d_in[6];
  const float* gf  = (const float*)d_in[7];
  const float* bef = (const float*)d_in[8];
  const float* Wd  = (const float*)d_in[9];
  const float* bd  = (const float*)d_in[10];
  const float* gd  = (const float*)d_in[11];
  const float* bed = (const float*)d_in[12];
  const float* Wu  = (const float*)d_in[13];
  const float* bu  = (const float*)d_in[14];
  const float* gu  = (const float*)d_in[15];
  const float* beu = (const float*)d_in[16];
  const float* gs  = (const float*)d_in[17];
  const float* bes = (const float*)d_in[18];
  const float* w1  = (const float*)d_in[19];
  const float* b1  = (const float*)d_in[20];
  const float* w2  = (const float*)d_in[21];
  const float* b2  = (const float*)d_in[22];
  const float* Wh  = (const float*)d_in[23];
  const float* bh  = (const float*)d_in[24];
  float* out = (float*)d_out;

  hipFuncSetAttribute((const void*)ctm_asr_kernel,
                      hipFuncAttributeMaxDynamicSharedMemorySize, SMEM_BYTES);
  ctm_asr_kernel<<<1, NT, SMEM_BYTES, stream>>>(
      X, st0, as0, Wb, bb, Wf, bf, gf, bef, Wd, bd, gd, bed,
      Wu, bu, gu, beu, gs, bes, w1, b1, w2, b2, Wh, bh, out);
}
